// AKDN_18966575579231
// MI455X (gfx1250) — matrix operand order repaired and checked by reading, not compiled, not run on hardware
//
#include <hip/hip_runtime.h>
#include <hip/hip_bf16.h>

#define N_ENT 100000
#define N_USR 30000
#define N_TOT 130000
#define N_REL 32
#define DIM   64
#define NEDGE 500000
#define NIG   1000000   // 2*INTER
#define SLOPE 0.01f

typedef __attribute__((ext_vector_type(16))) __bf16      v16bf;
typedef __attribute__((ext_vector_type(8)))  __bf16      v8bf;
typedef __attribute__((ext_vector_type(2)))  __bf16      v2bf;
typedef __attribute__((ext_vector_type(8)))  float       v8f;
typedef __attribute__((ext_vector_type(4)))  unsigned    u32x4;
typedef __attribute__((ext_vector_type(8)))  int         i32x8;
typedef __attribute__((ext_vector_type(4)))  int         i32x4;

#if defined(__HIP_DEVICE_COMPILE__) && __has_builtin(__builtin_amdgcn_tensor_load_to_lds) && __has_builtin(__builtin_amdgcn_s_wait_tensorcnt)
#define USE_TDM 1
#endif
#if defined(__HIP_DEVICE_COMPILE__) && __has_builtin(__builtin_amdgcn_perm)
#define USE_PERM 1
#endif

// round-to-nearest-even f32->bf16 (used in one-time packing/gather kernels)
__device__ __forceinline__ __bf16 f2bf(float f) {
    unsigned u = __float_as_uint(f);
    unsigned r = (u + 0x7FFFu + ((u >> 16) & 1u)) >> 16;
    unsigned short s = (unsigned short)r;
    return __builtin_bit_cast(__bf16, s);
}

// fast truncating pack of two f32 into two bf16 lanes: one v_perm_b32
__device__ __forceinline__ void put2(v16bf& v, int i, float x, float y) {
#ifdef USE_PERM
    unsigned r = __builtin_amdgcn_perm(__float_as_uint(y), __float_as_uint(x), 0x07060302u);
#else
    unsigned r = (__float_as_uint(x) >> 16) | (__float_as_uint(y) & 0xFFFF0000u);
#endif
    v2bf p = __builtin_bit_cast(v2bf, r);
    v[i] = p[0]; v[i + 1] = p[1];
}

// Stage 16KB (8192 bf16) from global into LDS via the Tensor Data Mover.
// One 1-row tile: data_size=2B, tensor_dim0 = tile_dim0 = 8192. Issued by wave 0.
__device__ __forceinline__ void tdm_stage_16kb(const __bf16* __restrict__ gsrc,
                                               __bf16* lds_dst) {
#ifdef USE_TDM
    if (threadIdx.x < 32) {
        unsigned long long ga = (unsigned long long)(const void*)gsrc;
        // generic LDS address: low 32 bits are the LDS byte offset (ISA 10.2 aperture rule)
        unsigned lo = (unsigned)(unsigned long long)(void*)lds_dst;
        u32x4 g0;
        g0[0] = 1u;                                        // count=1, user descriptor
        g0[1] = lo;                                        // lds_addr
        g0[2] = (unsigned)(ga & 0xFFFFFFFFull);            // global_addr[31:0]
        g0[3] = (unsigned)((ga >> 32) & 0x01FFFFFFull)     // global_addr[56:32]
              | (2u << 30);                                // type=2 ("image")
        const unsigned TD0 = 8192u, TD1 = 1u, TILE0 = 8192u, TILE1 = 1u;
        i32x8 g1;
        g1[0] = (int)(1u << 16);                           // data_size=1 -> 2 bytes
        g1[1] = (int)((TD0 & 0xFFFFu) << 16);              // tensor_dim0[15:0] @ bits63:48
        g1[2] = (int)(((TD0 >> 16) & 0xFFFFu) | ((TD1 & 0xFFFFu) << 16));
        g1[3] = (int)(((TD1 >> 16) & 0xFFFFu) | ((TILE0 & 0xFFFFu) << 16)); // tile_dim0
        g1[4] = (int)(TILE1 & 0xFFFFu);                    // tile_dim1=1, tile_dim2=0
        g1[5] = (int)TD0;                                  // tensor_dim0_stride[31:0]
        g1[6] = 0;
        g1[7] = 0;
        i32x4 gz4; gz4[0] = gz4[1] = gz4[2] = gz4[3] = 0;  // dims 3/4 unused
        i32x8 gz8;
        #pragma unroll
        for (int i = 0; i < 8; i++) gz8[i] = 0;
        __builtin_amdgcn_tensor_load_to_lds(g0, g1, gz4, gz4, gz8, 0);
        __builtin_amdgcn_s_wait_tensorcnt((short)0);
    }
    __syncthreads();
#else
    const float4* s = (const float4*)gsrc;
    float4* d = (float4*)lds_dst;
    for (int c = threadIdx.x; c < 1024; c += blockDim.x) d[c] = s[c];
    __syncthreads();
#endif
}

__global__ void zero_kernel(float* __restrict__ p, long n) {
    long i = (long)blockIdx.x * blockDim.x + threadIdx.x;
    if (i < n) p[i] = 0.0f;
}

// one-time RNE conversion of the embedding table to bf16 (halves gather traffic)
__global__ void conv_bf16_kernel(const float* __restrict__ src,
                                 __bf16* __restrict__ dst, long n) {
    long i = (long)blockIdx.x * blockDim.x + threadIdx.x;
    if (i < n) dst[i] = f2bf(src[i]);
}

// Pack Wk (64x128), Wa (64x64), Wb (64x64) into WMMA B-fragment layout:
// frag[(tile)*32 + lane][j] ; lane<16: col N=lane&15, K = kc*32 + hi*8 + (j&7) + (j>>3)*16
__global__ void pack_weights_kernel(const float* __restrict__ Wk,
                                    const float* __restrict__ Wa,
                                    const float* __restrict__ Wb,
                                    __bf16* __restrict__ WkB,
                                    __bf16* __restrict__ WaB,
                                    __bf16* __restrict__ WbB) {
    int idx = blockIdx.x * blockDim.x + threadIdx.x;
    if (idx < 8192) { // Wk: 4 n-tiles x 4 k-chunks x 32 lanes x 16
        int j = idx & 15, lane = (idx >> 4) & 31, fc = idx >> 9;
        int kc = fc & 3, nt = fc >> 2;
        int hi = lane >> 4, n = nt * 16 + (lane & 15);
        int k = kc * 32 + hi * 8 + (j & 7) + (j >> 3) * 16;
        WkB[idx] = f2bf(Wk[n * 128 + k]);
    } else if (idx < 16384) {
        int t = idx - 8192;
        int which = t >> 12;          // 0 = Wa, 1 = Wb
        int q = t & 4095;             // 4 n-tiles x 2 k-chunks x 32 x 16
        int j = q & 15, lane = (q >> 4) & 31, fc = q >> 9;
        int kc = fc & 1, nt = fc >> 1;
        int hi = lane >> 4, n = nt * 16 + (lane & 15);
        int k = kc * 32 + hi * 8 + (j & 7) + (j >> 3) * 16;
        const float* W = which ? Wb : Wa;
        __bf16* O = which ? WbB : WaB;
        O[q] = f2bf(W[n * 64 + k]);
    }
}

// Layer-1 edge attention: trans = [t_e|h_e] @ Wk^T + b ; logits = <trans, r_e>; leaky-relu.
// One wave per 16 edges; 4 N-tiles x 4 K-chunks of v_wmma_f32_16x16x32_bf16.
// A rows gathered directly in bf16; B fragments staged to LDS once per block by the TDM.
__global__ void edge_attn_kernel(const __bf16* __restrict__ allEmbBf,
                                 const float* __restrict__ relEmb,
                                 const __bf16* __restrict__ WkB,
                                 const float* __restrict__ Wk_b,
                                 const int* __restrict__ hL,
                                 const int* __restrict__ tL,
                                 const int* __restrict__ rL,
                                 float* __restrict__ vals) {
    __shared__ __align__(16) __bf16 sW[8192];
    tdm_stage_16kb(WkB, sW);                 // all waves reach the barrier inside

    int wid  = (blockIdx.x * blockDim.x + threadIdx.x) >> 5;
    int lane = threadIdx.x & 31;
    long eBase = (long)wid * 16;
    if (eBase >= NEDGE) return;
    int m = lane & 15, hi = lane >> 4;

    long eM = eBase + m; if (eM >= NEDGE) eM = NEDGE - 1;
    const __bf16* tRow = allEmbBf + (long)tL[eM] * DIM;
    const __bf16* hRow = allEmbBf + (long)hL[eM] * DIM;

    // A fragments for 4 K-chunks of the 16x128 cat tile (kc 0,1 -> t_e ; 2,3 -> h_e)
    v16bf a[4];
    #pragma unroll
    for (int kc = 0; kc < 4; kc++) {
        const __bf16* src = (kc < 2) ? tRow : hRow;
        int kb = (kc & 1) * 32 + hi * 8;
        v8bf lo = *(const v8bf*)(src + kb);          // K = kb..kb+7   (16B aligned)
        v8bf hh = *(const v8bf*)(src + kb + 16);     // K = kb+16..kb+23
        a[kc] = __builtin_shufflevector(lo, hh, 0, 1, 2, 3, 4, 5, 6, 7,
                                                 8, 9, 10, 11, 12, 13, 14, 15);
    }
    int ridx[8];
    #pragma unroll
    for (int g = 0; g < 8; g++) {
        long e = eBase + g + hi * 8; if (e >= NEDGE) e = NEDGE - 1;
        ridx[g] = rL[e];
    }

    float p[8];
    #pragma unroll
    for (int g = 0; g < 8; g++) p[g] = 0.0f;

    #pragma unroll
    for (int nt = 0; nt < 4; nt++) {
        v8f acc;
        #pragma unroll
        for (int i = 0; i < 8; i++) acc[i] = 0.0f;
        #pragma unroll
        for (int kc = 0; kc < 4; kc++) {
            v16bf b = *(const v16bf*)(sW + (((nt * 4 + kc) * 32 + lane) << 4));
            acc = __builtin_amdgcn_wmma_f32_16x16x32_bf16(false, a[kc], false, b,
                                                          (short)0, acc, false, false);
        }
        int n = nt * 16 + m;
        float bn = Wk_b[n];
        #pragma unroll
        for (int g = 0; g < 8; g++) {
            float rv = relEmb[(long)ridx[g] * DIM + n];
            p[g] += (acc[g] + bn) * rv;   // acc[g] = trans[row g+hi*8][col n]
        }
    }
    // sum over the 16 columns held by each half-wave
    #pragma unroll
    for (int off = 1; off < 16; off <<= 1)
        #pragma unroll
        for (int g = 0; g < 8; g++)
            p[g] += __shfl_xor(p[g], off, 32);

    if (m == 0) {
        #pragma unroll
        for (int g = 0; g < 8; g++) {
            long e = eBase + g + hi * 8;
            if (e < NEDGE) {
                float v = p[g];
                vals[e] = (v >= 0.0f) ? v : SLOPE * v;
            }
        }
    }
}

__global__ void seg_max_kernel(const float* __restrict__ vals,
                               const int* __restrict__ hL,
                               unsigned* __restrict__ keys) {
    long e = (long)blockIdx.x * blockDim.x + threadIdx.x;
    if (e >= NEDGE) return;
    unsigned u = __float_as_uint(vals[e]);
    unsigned key = (u & 0x80000000u) ? ~u : (u | 0x80000000u);
    atomicMax(&keys[hL[e]], key);
}

__global__ void seg_exp_kernel(float* __restrict__ vals,
                               const int* __restrict__ hL,
                               const unsigned* __restrict__ keys,
                               float* __restrict__ segSum) {
    long e = (long)blockIdx.x * blockDim.x + threadIdx.x;
    if (e >= NEDGE) return;
    int h = hL[e];
    unsigned key = keys[h];
    unsigned u = (key & 0x80000000u) ? (key & 0x7FFFFFFFu) : ~key;
    float ex = __expf(vals[e] - __uint_as_float(u));
    vals[e] = ex;
    atomicAdd(&segSum[h], ex);
}

__global__ void scatter_kg_kernel(const float* __restrict__ vals,
                                  const float* __restrict__ segSum,
                                  const int* __restrict__ hL,
                                  const int* __restrict__ tL,
                                  const float* __restrict__ allEmb,
                                  float* __restrict__ kg) {
    long idx = (long)blockIdx.x * blockDim.x + threadIdx.x;
    if (idx >= (long)NEDGE * DIM) return;
    long e = idx >> 6; int d = idx & 63;
    int h = hL[e];
    float alpha = vals[e] / segSum[h];
    atomicAdd(&kg[(long)h * DIM + d], alpha * allEmb[(long)tL[e] * DIM + d]);
}

// collab propagation: out[a_row] += a_vals * src[a_col]
__global__ void ig_prop_kernel(const float* __restrict__ itemsBase,
                               const float* __restrict__ usersBase,
                               const int* __restrict__ aRow,
                               const int* __restrict__ aCol,
                               const float* __restrict__ aVals,
                               float* __restrict__ out) {
    long idx = (long)blockIdx.x * blockDim.x + threadIdx.x;
    if (idx >= (long)NIG * DIM) return;
    long e = idx >> 6; int d = idx & 63;
    int col = aCol[e];
    float v = (col < N_ENT) ? itemsBase[(long)col * DIM + d]
                            : usersBase[(long)(col - N_ENT) * DIM + d];
    atomicAdd(&out[(long)aRow[e] * DIM + d], aVals[e] * v);
}

// gating: g = sigmoid(kg@Wa^T + collab@Wb^T); dual = g*kg + (1-g)*collab
// WaB||WbB (contiguous 16KB) staged to LDS by the TDM.
__global__ void gate_kernel(const float* __restrict__ kg,
                            const float* __restrict__ collab,
                            const __bf16* __restrict__ WaB,  // WbB = WaB + 4096 elements
                            float* __restrict__ dual) {
    __shared__ __align__(16) __bf16 sW[8192];
    tdm_stage_16kb(WaB, sW);

    int wid  = (blockIdx.x * blockDim.x + threadIdx.x) >> 5;
    int lane = threadIdx.x & 31;
    long rBase = (long)wid * 16;
    if (rBase >= N_ENT) return;
    int m = lane & 15, hi = lane >> 4;
    long row = rBase + m; if (row >= N_ENT) row = N_ENT - 1;

    v16bf aK[2], aC[2];
    #pragma unroll
    for (int kc = 0; kc < 2; kc++) {
        int kb = kc * 32 + hi * 8;
        #pragma unroll
        for (int j = 0; j < 8; j += 2) {
            put2(aK[kc], j,     kg[row * DIM + kb + j],          kg[row * DIM + kb + j + 1]);
            put2(aK[kc], 8 + j, kg[row * DIM + kb + 16 + j],     kg[row * DIM + kb + 17 + j]);
            put2(aC[kc], j,     collab[row * DIM + kb + j],      collab[row * DIM + kb + j + 1]);
            put2(aC[kc], 8 + j, collab[row * DIM + kb + 16 + j], collab[row * DIM + kb + 17 + j]);
        }
    }
    #pragma unroll
    for (int nt = 0; nt < 4; nt++) {
        v8f accA, accB;
        #pragma unroll
        for (int i = 0; i < 8; i++) { accA[i] = 0.0f; accB[i] = 0.0f; }
        #pragma unroll
        for (int kc = 0; kc < 2; kc++) {
            v16bf bA = *(const v16bf*)(sW +        (((nt * 2 + kc) * 32 + lane) << 4));
            v16bf bB = *(const v16bf*)(sW + 4096 + (((nt * 2 + kc) * 32 + lane) << 4));
            accA = __builtin_amdgcn_wmma_f32_16x16x32_bf16(false, aK[kc], false, bA,
                                                           (short)0, accA, false, false);
            accB = __builtin_amdgcn_wmma_f32_16x16x32_bf16(false, aC[kc], false, bB,
                                                           (short)0, accB, false, false);
        }
        int n = nt * 16 + m;
        #pragma unroll
        for (int g = 0; g < 8; g++) {
            long rr = rBase + g + hi * 8;
            if (rr < N_ENT) {
                float kgv = kg[rr * DIM + n];
                float cbv = collab[rr * DIM + n];
                float s = 1.0f / (1.0f + __expf(-(accA[g] + accB[g])));
                dual[rr * DIM + n] = s * kgv + (1.0f - s) * cbv;
            }
        }
    }
}

// all_final[i] = all_embed[i] + ig1[i] + ig2[i]; gather user/item rows to bf16
__global__ void gather_final_kernel(const float* __restrict__ allEmb,
                                    const float* __restrict__ ig1,
                                    const float* __restrict__ ig2,
                                    const int* __restrict__ userIds,
                                    const int* __restrict__ itemIds,
                                    __bf16* __restrict__ Ubf,
                                    __bf16* __restrict__ Ibf) {
    long idx = (long)blockIdx.x * blockDim.x + threadIdx.x;
    if (idx >= (long)(1024 + 2048) * DIM) return;
    if (idx < (long)1024 * DIM) {
        long r = idx >> 6; int d = idx & 63;
        long id = userIds[r];
        Ubf[idx] = f2bf(allEmb[id * DIM + d] + ig1[id * DIM + d] + ig2[id * DIM + d]);
    } else {
        long t = idx - (long)1024 * DIM;
        long r = t >> 6; int d = t & 63;
        long id = itemIds[r];
        Ibf[t] = f2bf(allEmb[id * DIM + d] + ig1[id * DIM + d] + ig2[id * DIM + d]);
    }
}

// out (1024x2048) = U (1024x64) @ I^T (64x2048); one wave per 16x16 tile
__global__ void score_gemm_kernel(const __bf16* __restrict__ Ubf,
                                  const __bf16* __restrict__ Ibf,
                                  float* __restrict__ out) {
    int wid  = (blockIdx.x * blockDim.x + threadIdx.x) >> 5;
    int lane = threadIdx.x & 31;
    int mt = wid >> 7, ntile = wid & 127;       // 64 x 128 tiles
    if (mt >= 64) return;
    int mBase = mt * 16, nBase = ntile * 16;
    int m = lane & 15, hi = lane >> 4;

    v8f acc;
    #pragma unroll
    for (int i = 0; i < 8; i++) acc[i] = 0.0f;
    #pragma unroll
    for (int kc = 0; kc < 2; kc++) {
        const __bf16* ar = Ubf + (long)(mBase + m) * DIM + kc * 32 + hi * 16;
        v8bf lo = *(const v8bf*)ar;
        v8bf hh = *(const v8bf*)(ar + 8);
        v16bf a = __builtin_shufflevector(lo, hh, 0, 1, 2, 3, 4, 5, 6, 7,
                                                   8, 9, 10, 11, 12, 13, 14, 15);
        v16bf b = *(const v16bf*)(Ibf + (long)(nBase + m) * DIM + kc * 32 + hi * 16);
        acc = __builtin_amdgcn_wmma_f32_16x16x32_bf16(false, a, false, b,
                                                      (short)0, acc, false, false);
    }
    #pragma unroll
    for (int g = 0; g < 8; g++)
        out[(long)(mBase + g + hi * 8) * 2048 + nBase + m] = acc[g];
}

extern "C" void kernel_launch(void* const* d_in, const int* in_sizes, int n_in,
                              void* d_out, int out_size, void* d_ws, size_t ws_size,
                              hipStream_t stream) {
    const float* allEmb  = (const float*)d_in[0];
    const float* relEmb  = (const float*)d_in[1];
    const float* Wk_w    = (const float*)d_in[2];
    const float* Wk_b    = (const float*)d_in[3];
    const float* Wa_w    = (const float*)d_in[4];
    const float* Wb_w    = (const float*)d_in[5];
    const float* a_vals  = (const float*)d_in[6];
    const int*   userIds = (const int*)d_in[7];
    const int*   itemIds = (const int*)d_in[8];
    const int*   hL      = (const int*)d_in[9];
    const int*   tL      = (const int*)d_in[10];
    const int*   rL      = (const int*)d_in[11];
    const int*   aRow    = (const int*)d_in[12];
    const int*   aCol    = (const int*)d_in[13];
    float* out = (float*)d_out;

    char* w = (char*)d_ws;
    auto carve = [&](size_t bytes) -> void* {
        void* p = (void*)w;
        w += (bytes + 255) & ~(size_t)255;
        return p;
    };
    float*    ig1      = (float*)carve((size_t)N_TOT * DIM * 4);
    float*    ig2      = (float*)carve((size_t)N_TOT * DIM * 4);
    float*    kg1      = (float*)carve((size_t)N_ENT * DIM * 4);
    float*    dual1    = (float*)carve((size_t)N_ENT * DIM * 4);
    float*    vals     = (float*)carve((size_t)NEDGE * 4);
    unsigned* keys     = (unsigned*)carve((size_t)N_ENT * 4);
    float*    segSum   = (float*)carve((size_t)N_ENT * 4);
    __bf16*   allEmbBf = (__bf16*)carve((size_t)N_TOT * DIM * 2);
    __bf16*   WkB      = (__bf16*)carve((size_t)8192 * 2);
    // WaB and WbB MUST stay contiguous (one 16KB TDM stage covers both)
    __bf16*   WaB      = (__bf16*)carve((size_t)8192 * 2);
    __bf16*   WbB      = WaB + 4096;
    __bf16*   Ubf      = (__bf16*)carve((size_t)1024 * DIM * 2);
    __bf16*   Ibf      = (__bf16*)carve((size_t)2048 * DIM * 2);

    const int B = 256;
    auto gridFor = [](long n, int b) { return (int)((n + b - 1) / b); };

    pack_weights_kernel<<<64, B, 0, stream>>>(Wk_w, Wa_w, Wb_w, WkB, WaB, WbB);
    conv_bf16_kernel<<<gridFor((long)N_TOT * DIM, B), B, 0, stream>>>(
        allEmb, allEmbBf, (long)N_TOT * DIM);

    zero_kernel<<<gridFor((long)N_TOT * DIM, B), B, 0, stream>>>(ig1, (long)N_TOT * DIM);
    zero_kernel<<<gridFor((long)N_TOT * DIM, B), B, 0, stream>>>(ig2, (long)N_TOT * DIM);
    zero_kernel<<<gridFor((long)N_ENT * DIM, B), B, 0, stream>>>(kg1, (long)N_ENT * DIM);
    zero_kernel<<<gridFor(N_ENT, B), B, 0, stream>>>((float*)keys, N_ENT);
    zero_kernel<<<gridFor(N_ENT, B), B, 0, stream>>>(segSum, N_ENT);

    // Layer 1 KG attention (only layer whose attention output is live)
    {
        long waves = (NEDGE + 15) / 16;
        edge_attn_kernel<<<gridFor(waves * 32, B), B, 0, stream>>>(
            allEmbBf, relEmb, WkB, Wk_b, hL, tL, rL, vals);
    }
    seg_max_kernel<<<gridFor(NEDGE, B), B, 0, stream>>>(vals, hL, keys);
    seg_exp_kernel<<<gridFor(NEDGE, B), B, 0, stream>>>(vals, hL, keys, segSum);
    scatter_kg_kernel<<<gridFor((long)NEDGE * DIM, B), B, 0, stream>>>(
        vals, segSum, hL, tL, allEmb, kg1);

    // Layer-1 collab propagation: src = all_embed
    ig_prop_kernel<<<gridFor((long)NIG * DIM, B), B, 0, stream>>>(
        allEmb, allEmb + (long)N_ENT * DIM, aRow, aCol, a_vals, ig1);

    // Gating -> dual1 (input to layer-2 propagation)
    {
        long waves = (N_ENT + 15) / 16;
        gate_kernel<<<gridFor(waves * 32, B), B, 0, stream>>>(kg1, ig1, WaB, dual1);
    }

    // Layer-2 collab propagation: items = dual1, users = ig1 users-part
    ig_prop_kernel<<<gridFor((long)NIG * DIM, B), B, 0, stream>>>(
        dual1, ig1 + (long)N_ENT * DIM, aRow, aCol, a_vals, ig2);

    gather_final_kernel<<<gridFor((long)(1024 + 2048) * DIM, B), B, 0, stream>>>(
        allEmb, ig1, ig2, userIds, itemIds, Ubf, Ibf);

    score_gemm_kernel<<<gridFor((long)64 * 128 * 32, B), B, 0, stream>>>(Ubf, Ibf, out);
}
